// ErrGAT_13108240187633
// MI455X (gfx1250) — compile-verified, matrix operand order checked by reading, not run on hardware
//
#include <hip/hip_runtime.h>
#include <stdint.h>

// Problem constants (match reference)
#define NN   50000
#define EE   800000
#define F_IN 64
#define HID  128
#define NH   4
#define CC   32
#define GG   64
#define HALF 64
#define NEG_SLOPE 0.2f

typedef __bf16 bf16_t;
typedef __attribute__((ext_vector_type(16))) bf16_t v16bf;
typedef __attribute__((ext_vector_type(8)))  float  v8f;

// ---------------- helpers ----------------
__device__ __forceinline__ unsigned short f2bf(float f) {
    unsigned b = __float_as_uint(f);
    unsigned r = b + 0x7FFFu + ((b >> 16) & 1u);   // round-to-nearest-even
    return (unsigned short)(r >> 16);
}
// order-preserving float<->uint encoding for atomicMax on signed floats
__device__ __forceinline__ unsigned encf(float f) {
    unsigned b = __float_as_uint(f);
    return (b & 0x80000000u) ? ~b : (b | 0x80000000u);
}
__device__ __forceinline__ float decf(unsigned k) {
    unsigned b = (k & 0x80000000u) ? (k ^ 0x80000000u) : ~k;
    return __uint_as_float(b);
}

// ---------------- fill kernels ----------------
__global__ void gat_fill_f32(float* p, long n, float v) {
    long i = (long)blockIdx.x * blockDim.x + threadIdx.x;
    if (i < n) p[i] = v;
}
__global__ void gat_fill_u32(unsigned* p, long n, unsigned v) {
    long i = (long)blockIdx.x * blockDim.x + threadIdx.x;
    if (i < n) p[i] = v;
}
__global__ void gat_cvt_bf16(const float* __restrict__ in, unsigned short* __restrict__ out, long n) {
    long i = (long)blockIdx.x * blockDim.x + threadIdx.x;
    if (i < n) out[i] = f2bf(in[i]);
}

// ---------------- pack W (fp32 [K][128]) into WMMA B fragments ----------------
// Bpack[((ntile*KT + kt)*32 + lane)*8 + v] = u32 of {bf16 W[k][n], bf16 W[k+1][n]}
// lane: n = ntile*16 + (lane&15); hi = lane>>4
// v:    kbase = (v<4 ? 2v : 16+2(v-4)) + 8*hi;  k = kt*32 + kbase
__global__ void gat_pack_w(const float* __restrict__ W, unsigned* __restrict__ bp, int Kdim) {
    int KT = Kdim >> 5;
    int total = 8 * KT * 32 * 8;
    int idx = blockIdx.x * blockDim.x + threadIdx.x;
    if (idx >= total) return;
    int v    = idx & 7;
    int lane = (idx >> 3) & 31;
    int rest = idx >> 8;
    int kt   = rest % KT;
    int nt   = rest / KT;
    int n  = nt * 16 + (lane & 15);
    int hi = lane >> 4;
    int kbase = ((v < 4) ? (2 * v) : (16 + 2 * (v - 4))) + 8 * hi;
    int k = kt * 32 + kbase;
    unsigned lo = f2bf(W[(size_t)k * HID + n]);
    unsigned hi16 = f2bf(W[(size_t)(k + 1) * HID + n]);
    bp[idx] = lo | (hi16 << 16);
}

// ---------------- WMMA GEMM:  out[N][128] = A_bf16[N][Kdim] @ W + bias ----------------
// block = 256 threads = 8 waves; wave w computes col tile w (cols 16w..16w+15)
// grid.x = N/16 row tiles (N = 50000 = 16*3125 exactly)
__global__ void gat_gemm_wmma(const unsigned short* __restrict__ A,
                              const unsigned* __restrict__ Bp,
                              const float* __restrict__ bias,
                              float* __restrict__ out, int Kdim) {
    int lane = threadIdx.x & 31;
    int w    = threadIdx.x >> 5;           // 0..7 : column tile
    int mt   = blockIdx.x;                 // row tile
    int hi   = lane >> 4;
    int lo16 = lane & 15;
    int KT   = Kdim >> 5;

    int row = mt * 16 + lo16;              // A-matrix: lanes 0-15 & 16-31 both map M=lane&15
    const unsigned short* arow = A + (size_t)row * Kdim;

    v8f acc = {};
    for (int kt = 0; kt < KT; ++kt) {
        int k0 = kt * 32;
        union { uint4 q[2]; v16bf v; } ua, ub;
        ua.q[0] = *(const uint4*)(arow + k0 + 8 * hi);        // K = k0+8hi .. +7
        ua.q[1] = *(const uint4*)(arow + k0 + 16 + 8 * hi);   // K = k0+16+8hi .. +7
        const uint4* pb = (const uint4*)(Bp + ((size_t)(w * KT + kt) * 32 + lane) * 8);
        ub.q[0] = pb[0];
        ub.q[1] = pb[1];
        acc = __builtin_amdgcn_wmma_f32_16x16x32_bf16(false, ua.v, false, ub.v,
                                                      (short)0, acc, false, false);
    }
    int col = w * 16 + lo16;
    float b = bias[col];
    float* orow = out + (size_t)(mt * 16 + 8 * hi) * HID + col;
#pragma unroll
    for (int r = 0; r < 8; ++r)
        orow[(size_t)r * HID] = acc[r] + b;                    // M = r + 8*hi
}

// ---------------- edge: logits + segment max ----------------
__global__ void gat_edge_logits(const int* __restrict__ src, const int* __restrict__ dst,
                                const float* __restrict__ xl, const float* __restrict__ xr,
                                const float* __restrict__ att,
                                float* __restrict__ logits, unsigned* __restrict__ menc) {
    int idx = blockIdx.x * blockDim.x + threadIdx.x;
    if (idx >= EE * NH) return;
    int e = idx >> 2, h = idx & 3;
    int s = src[e], d = dst[e];
    const float4* pl = (const float4*)(xl + (size_t)s * HID + h * CC);
    const float4* pr = (const float4*)(xr + (size_t)d * HID + h * CC);
    const float4* pa = (const float4*)(att + h * CC);
    float acc = 0.f;
#pragma unroll
    for (int i = 0; i < 8; ++i) {
        float4 a = pl[i], b = pr[i], t = pa[i];
        float v;
        v = a.x + b.x; v = (v > 0.f) ? v : NEG_SLOPE * v; acc += v * t.x;
        v = a.y + b.y; v = (v > 0.f) ? v : NEG_SLOPE * v; acc += v * t.y;
        v = a.z + b.z; v = (v > 0.f) ? v : NEG_SLOPE * v; acc += v * t.z;
        v = a.w + b.w; v = (v > 0.f) ? v : NEG_SLOPE * v; acc += v * t.w;
    }
    logits[idx] = acc;
    atomicMax(&menc[(size_t)d * NH + h], encf(acc));
}

// ---------------- edge: p = exp(logit - max), denom += p ----------------
__global__ void gat_edge_expsum(const int* __restrict__ dst,
                                float* __restrict__ logits,   // in: logits, out: p
                                const unsigned* __restrict__ menc,
                                float* __restrict__ denom) {
    int idx = blockIdx.x * blockDim.x + threadIdx.x;
    if (idx >= EE * NH) return;
    int e = idx >> 2, h = idx & 3;
    int d = dst[e];
    float m = decf(menc[(size_t)d * NH + h]);
    float p = __expf(logits[idx] - m);
    logits[idx] = p;
    atomicAdd(&denom[(size_t)d * NH + h], p);
}

// ---------------- edge: out[dst] += xl[src] * alpha ----------------
__global__ void gat_edge_aggregate(const int* __restrict__ src, const int* __restrict__ dst,
                                   const float* __restrict__ xl,
                                   const float* __restrict__ p,
                                   const float* __restrict__ denom,
                                   float* __restrict__ agg) {
    int idx = blockIdx.x * blockDim.x + threadIdx.x;
    if (idx >= EE * NH) return;
    int e = idx >> 2, h = idx & 3;
    int s = src[e], d = dst[e];
    float alpha = p[idx] / (denom[(size_t)d * NH + h] + 1e-16f);
    const float4* pl = (const float4*)(xl + (size_t)s * HID + h * CC);
    float* po = agg + (size_t)d * HID + h * CC;
#pragma unroll
    for (int i = 0; i < 8; ++i) {
        float4 v = pl[i];
        atomicAdd(po + 4 * i + 0, v.x * alpha);
        atomicAdd(po + 4 * i + 1, v.y * alpha);
        atomicAdd(po + 4 * i + 2, v.z * alpha);
        atomicAdd(po + 4 * i + 3, v.w * alpha);
    }
}

// ---------------- bias + relu + pack to bf16 for next layer ----------------
__global__ void gat_finalize_relu(const float* __restrict__ agg, const float* __restrict__ bout,
                                  unsigned short* __restrict__ hb) {
    long i = (long)blockIdx.x * blockDim.x + threadIdx.x;
    if (i >= (long)NN * HID) return;
    float v = agg[i] + bout[(int)(i & (HID - 1))];
    v = (v > 0.f) ? v : 0.f;
    hb[i] = f2bf(v);
}

// ---------------- global mean pool (sum + count) ----------------
__global__ void gat_pool(const float* __restrict__ agg, const float* __restrict__ bout,
                         const int* __restrict__ batch,
                         float* __restrict__ sums, float* __restrict__ counts) {
    long i = (long)blockIdx.x * blockDim.x + threadIdx.x;
    if (i >= (long)NN * HID) return;
    int node = (int)(i >> 7);
    int c = (int)(i & (HID - 1));
    int g = batch[node];
    atomicAdd(&sums[(size_t)g * HID + c], agg[i] + bout[c]);
    if (c == 0) atomicAdd(&counts[g], 1.0f);
}

// ---------------- MLP head: relu(pool@W1+b1)@W2+b2 -> sigmoid ----------------
__global__ void gat_mlp_head(const float* __restrict__ sums, const float* __restrict__ counts,
                             const float* __restrict__ W1, const float* __restrict__ b1,
                             const float* __restrict__ W2, const float* __restrict__ b2,
                             float* __restrict__ out) {
    int g = threadIdx.x;
    if (g >= GG) return;
    float cnt = counts[g];
    if (cnt < 1.0f) cnt = 1.0f;
    float inv = 1.0f / cnt;
    float z = 0.f;
    for (int j = 0; j < HALF; ++j) {
        float s = b1[j];
        for (int k = 0; k < HID; ++k)
            s += (sums[(size_t)g * HID + k] * inv) * W1[(size_t)k * HALF + j];
        s = (s > 0.f) ? s : 0.f;
        z += s * W2[j];
    }
    z += b2[0];
    out[g] = 1.0f / (1.0f + __expf(-z));
}

// ---------------- host ----------------
static inline size_t bump(size_t& off, size_t bytes) {
    size_t r = off;
    off = (off + bytes + 255) & ~(size_t)255;
    return r;
}

extern "C" void kernel_launch(void* const* d_in, const int* in_sizes, int n_in,
                              void* d_out, int out_size, void* d_ws, size_t ws_size,
                              hipStream_t stream) {
    (void)in_sizes; (void)n_in; (void)out_size; (void)ws_size;
    const float* x      = (const float*)d_in[0];
    const int*   ei     = (const int*)d_in[1];
    const int*   batch  = (const int*)d_in[2];
    const int*   srcp   = ei;
    const int*   dstp   = ei + EE;
    float* out = (float*)d_out;

    char* ws = (char*)d_ws;
    size_t off = 0;
    unsigned short* hb   = (unsigned short*)(ws + bump(off, (size_t)NN * HID * 2));
    unsigned* bsrcp      = (unsigned*)(ws + bump(off, 8 * 4 * 32 * 8 * 4));
    unsigned* bdstp      = (unsigned*)(ws + bump(off, 8 * 4 * 32 * 8 * 4));
    float* xl            = (float*)(ws + bump(off, (size_t)NN * HID * 4));
    float* xr            = (float*)(ws + bump(off, (size_t)NN * HID * 4));
    float* logits        = (float*)(ws + bump(off, (size_t)EE * NH * 4));
    unsigned* menc       = (unsigned*)(ws + bump(off, (size_t)NN * NH * 4));
    float* denom         = (float*)(ws + bump(off, (size_t)NN * NH * 4));
    float* agg           = (float*)(ws + bump(off, (size_t)NN * HID * 4));
    float* sums          = (float*)(ws + bump(off, (size_t)GG * HID * 4));
    float* counts        = (float*)(ws + bump(off, (size_t)GG * 4));

    const int BT = 256;
    const long nNH = (long)NN * HID;
    const int  ehBlocks = (EE * NH + BT - 1) / BT;

    // layer 0 input -> bf16
    gat_cvt_bf16<<<(int)(((long)NN * F_IN + BT - 1) / BT), BT, 0, stream>>>(x, hb, (long)NN * F_IN);

    for (int l = 0; l < 3; ++l) {
        int Kdim = (l == 0) ? F_IN : HID;
        int KT = Kdim >> 5;
        const float* Wsrc = (const float*)d_in[3 + l];
        const float* bsrc = (const float*)d_in[6 + l];
        const float* Wdst = (const float*)d_in[9 + l];
        const float* bdst = (const float*)d_in[12 + l];
        const float* att  = (const float*)d_in[15 + l];
        const float* bout = (const float*)d_in[18 + l];

        int packThreads = 8 * KT * 32 * 8;
        gat_pack_w<<<(packThreads + BT - 1) / BT, BT, 0, stream>>>(Wsrc, bsrcp, Kdim);
        gat_pack_w<<<(packThreads + BT - 1) / BT, BT, 0, stream>>>(Wdst, bdstp, Kdim);

        gat_gemm_wmma<<<NN / 16, 256, 0, stream>>>(hb, bsrcp, bsrc, xl, Kdim);
        gat_gemm_wmma<<<NN / 16, 256, 0, stream>>>(hb, bdstp, bdst, xr, Kdim);

        gat_fill_f32<<<(int)((nNH + BT - 1) / BT), BT, 0, stream>>>(agg, nNH, 0.f);
        gat_fill_f32<<<((NN * NH) + BT - 1) / BT, BT, 0, stream>>>(denom, (long)NN * NH, 0.f);
        gat_fill_u32<<<((NN * NH) + BT - 1) / BT, BT, 0, stream>>>(menc, (long)NN * NH, 0x007FFFFFu); // enc(-inf)

        gat_edge_logits<<<ehBlocks, BT, 0, stream>>>(srcp, dstp, xl, xr, att, logits, menc);
        gat_edge_expsum<<<ehBlocks, BT, 0, stream>>>(dstp, logits, menc, denom);
        gat_edge_aggregate<<<ehBlocks, BT, 0, stream>>>(srcp, dstp, xl, logits, denom, agg);

        if (l < 2) {
            gat_finalize_relu<<<(int)((nNH + BT - 1) / BT), BT, 0, stream>>>(agg, bout, hb);
        } else {
            gat_fill_f32<<<((GG * HID) + BT - 1) / BT, BT, 0, stream>>>(sums, (long)GG * HID, 0.f);
            gat_fill_f32<<<1, GG, 0, stream>>>(counts, GG, 0.f);
            gat_pool<<<(int)((nNH + BT - 1) / BT), BT, 0, stream>>>(agg, bout, batch, sums, counts);
            gat_mlp_head<<<1, GG, 0, stream>>>(sums, counts,
                                               (const float*)d_in[21], (const float*)d_in[22],
                                               (const float*)d_in[23], (const float*)d_in[24],
                                               out);
        }
    }
}
